// AxialMultiAttention_223338300230
// MI455X (gfx1250) — matrix operand order repaired and checked by reading, not compiled, not run on hardware
//
#include <hip/hip_runtime.h>
#include <hip/hip_bf16.h>
#include <math.h>

#define BS   4
#define MDIM 190
#define TDIM 512
#define CDIM 64
#define NH   4
#define DH   16

#define QT 4   // q-tiles (16 rows each) per block
#define MT 2   // m values per block

typedef __attribute__((ext_vector_type(16))) __bf16 v16bf;
typedef __attribute__((ext_vector_type(8)))  float  v8f;

__device__ __forceinline__ unsigned short f2bf(float f) {
  union { float f; unsigned u; } x; x.f = f;
  unsigned r = x.u + 0x7FFFu + ((x.u >> 16) & 1u);   // round-to-nearest-even
  return (unsigned short)(r >> 16);
}

// ---------------------------------------------------------------------------
// K0: pack Wp (64x64 f32, row=k, col=c) into WMMA B-layout bf16 tiles.
// Tile (j = col/16, ki = k/32): lane = (col&15) + 16*(kk>=16), e = kk&15.
__global__ void pack_wp_kernel(const float* __restrict__ Wp,
                               unsigned short* __restrict__ wpP) {
  int idx = blockIdx.x * blockDim.x + threadIdx.x;
  if (idx >= 64 * 64) return;
  int k = idx >> 6, col = idx & 63;
  int j = col >> 4, ct = col & 15;
  int ki = k >> 5, kk = k & 31;
  int lane = ct + ((kk & 16) ? 16 : 0);
  int e = kk & 15;
  wpP[(((j * 2 + ki) * 32 + lane) << 4) + e] = f2bf(Wp[idx]);
}

// ---------------------------------------------------------------------------
// K1: xs[b,t,c] = sum_m x[b,m,t,c]
__global__ void reduce_x_kernel(const float* __restrict__ x,
                                float* __restrict__ xs) {
  int idx = blockIdx.x * blockDim.x + threadIdx.x;   // Bs*T*C = 131072
  int b = idx / (TDIM * CDIM);
  int tc = idx - b * (TDIM * CDIM);
  const float* p = x + (size_t)b * MDIM * (TDIM * CDIM) + tc;
  float s = 0.f;
  for (int m = 0; m < MDIM; ++m) s += p[(size_t)m * (TDIM * CDIM)];
  xs[idx] = s;
}

// ---------------------------------------------------------------------------
// K2: qt/kt in head-major layout [b][n][t][16]; qt = xs@Wq + M*bq, kt = xs@Wk + M*bk
__global__ void qtkt_kernel(const float* __restrict__ xs,
                            const float* __restrict__ Wq, const float* __restrict__ bq,
                            const float* __restrict__ Wk, const float* __restrict__ bk,
                            float* __restrict__ qt, float* __restrict__ kt) {
  int idx = blockIdx.x * blockDim.x + threadIdx.x;   // Bs*NH*T*DH = 131072
  int d = idx & 15;
  int t = (idx >> 4) & 511;
  int n = (idx >> 13) & 3;
  int b = idx >> 15;
  int c = n * 16 + d;
  const float* xrow = xs + (((size_t)b * TDIM + t) << 6);
  float sq = 0.f, sk = 0.f;
  for (int kc = 0; kc < 64; ++kc) {
    float xv = xrow[kc];
    sq += xv * Wq[kc * 64 + c];
    sk += xv * Wk[kc * 64 + c];
  }
  qt[idx] = sq + (float)MDIM * bq[c];
  kt[idx] = sk + (float)MDIM * bk[c];
}

// ---------------------------------------------------------------------------
// K3: v = x@Wv + bv, packed directly into WMMA B-layout bf16:
// vP[(((bm*NH+n)*16 + ki)*32 + lane)*16 + e], lane = d + 16*(kk>=16), e = kk&15
__global__ __launch_bounds__(256)
void vproj_pack_kernel(const float* __restrict__ x,
                       const float* __restrict__ Wv, const float* __restrict__ bv,
                       unsigned short* __restrict__ vP) {
  __shared__ float xrow[4][64];
  int r = threadIdx.x >> 6;
  int c = threadIdx.x & 63;
  size_t row = (size_t)blockIdx.x * 4 + r;           // index in [0, Bs*M*T)
  xrow[r][c] = x[row * 64 + c];
  __syncthreads();
  float s = 0.f;
  for (int kc = 0; kc < 64; ++kc) s += xrow[r][kc] * Wv[kc * 64 + c];
  s += bv[c];
  int t = (int)(row & 511);
  size_t bm = row >> 9;                              // b*M + m
  int n = c >> 4, d = c & 15;
  int ki = t >> 5, kk = t & 31;
  int lane = d + ((kk & 16) ? 16 : 0);
  int e = kk & 15;
  vP[(((((bm * NH + n) * 16 + ki) * 32 + lane)) << 4) + e] = f2bf(s);
}

// ---------------------------------------------------------------------------
// K4: scores + softmax. One block per (b,n,q). Writes packed-A bf16 weights,
// and f32 'a' output for b==0.
__global__ __launch_bounds__(128)
void softmax_kernel(const float* __restrict__ qt, const float* __restrict__ kt,
                    unsigned short* __restrict__ wP, float* __restrict__ aout) {
  __shared__ float ktile[TDIM * DH];                 // 32 KB
  __shared__ float red[128];
  int q = blockIdx.x, n = blockIdx.y, b = blockIdx.z;
  int tid = threadIdx.x;
  const float* ktBase = kt + (((size_t)(b * NH + n) * TDIM) << 4);
  for (int i = tid; i < TDIM * DH / 4; i += 128)
    ((float4*)ktile)[i] = ((const float4*)ktBase)[i];
  float qrow[16];
  const float* qp = qt + ((((size_t)(b * NH + n) * TDIM) + q) << 4);
#pragma unroll
  for (int d = 0; d < 16; ++d) qrow[d] = qp[d];
  __syncthreads();

  const float scale = rsqrtf((float)(DH * MDIM));
  float s[4];
  float mx = -1e30f;
#pragma unroll
  for (int i = 0; i < 4; ++i) {
    int k = tid + i * 128;
    const float* kr = ktile + k * 16;
    float acc = 0.f;
#pragma unroll
    for (int d = 0; d < 16; ++d) acc += qrow[d] * kr[d];
    acc *= scale;
    s[i] = acc;
    mx = fmaxf(mx, acc);
  }
  red[tid] = mx; __syncthreads();
  for (int st = 64; st > 0; st >>= 1) {
    if (tid < st) red[tid] = fmaxf(red[tid], red[tid + st]);
    __syncthreads();
  }
  mx = red[0]; __syncthreads();
  float sum = 0.f;
#pragma unroll
  for (int i = 0; i < 4; ++i) { s[i] = __expf(s[i] - mx); sum += s[i]; }
  red[tid] = sum; __syncthreads();
  for (int st = 64; st > 0; st >>= 1) {
    if (tid < st) red[tid] += red[tid + st];
    __syncthreads();
  }
  float inv = 1.0f / red[0];

  int qi = q >> 4, L0 = q & 15;
  size_t wqbase = (size_t)((b * NH + n) * 32 + qi) * 8192;   // *16 ki *32 lane *16 e
#pragma unroll
  for (int i = 0; i < 4; ++i) {
    int k = tid + i * 128;
    float w = s[i] * inv;
    int ki = k >> 5, kk = k & 31;
    int lane = L0 + ((kk & 16) ? 16 : 0);
    int e = kk & 15;
    wP[wqbase + (size_t)(((ki * 32 + lane) << 4) + e)] = f2bf(w);
    if (b == 0) aout[((size_t)n * TDIM + q) * TDIM + k] = w;
  }
}

// ---------------------------------------------------------------------------
// K5: heavy einsum + output projection, all WMMA, register-blocked QT x MT,
// with explicit double-buffered software pipelining in the K-loop:
// two operand buffer sets alternate across a 2x-unrolled loop so the loads
// for step ki+1 are in flight while the 8 WMMAs of step ki execute.
__global__ __launch_bounds__(128)
void attn_wmma_kernel(const unsigned short* __restrict__ wP,
                      const unsigned short* __restrict__ vP,
                      const unsigned short* __restrict__ wpP,
                      const float* __restrict__ bp,
                      float* __restrict__ out) {
  __shared__ unsigned short att[MT][QT][16 * 68];    // 17 KB
  int qb = blockIdx.x;                               // 0..(T/(16*QT))-1
  int m0 = blockIdx.y * MT;
  int b  = blockIdx.z;
  int wave = threadIdx.x >> 5;
  int lane = threadIdx.x & 31;
  int hi = lane >> 4;
  int lr = lane & 15;

  // ---- Stage 1: att[mm][t] = W[b,n][qtile t] @ V[b,m0+mm,n], K = 512 ----
  int n = wave;
  const unsigned short* aBase =
      wP + (size_t)((b * NH + n) * 32 + qb * QT) * 8192 + (lane << 4);
  const unsigned short* vBase =
      vP + (size_t)((b * MDIM + m0) * NH + n) * 8192 + (lane << 4);

  v8f acc[MT][QT] = {};
  v16bf aA[QT], bA[MT], aB[QT], bB[MT];

  // prologue: load ki = 0 into A buffers
#pragma unroll
  for (int mm = 0; mm < MT; ++mm)
    bA[mm] = *(const v16bf*)(vBase + (size_t)mm * (NH * 8192));
#pragma unroll
  for (int t = 0; t < QT; ++t)
    aA[t] = *(const v16bf*)(aBase + (size_t)t * 8192);

  for (int ki = 0; ki < 16; ki += 2) {
    // prefetch ki+1 into B buffers
#pragma unroll
    for (int mm = 0; mm < MT; ++mm)
      bB[mm] = *(const v16bf*)(vBase + (size_t)mm * (NH * 8192) + (ki + 1) * 512);
#pragma unroll
    for (int t = 0; t < QT; ++t)
      aB[t] = *(const v16bf*)(aBase + (size_t)t * 8192 + (ki + 1) * 512);
    // consume A buffers (ki)
#pragma unroll
    for (int t = 0; t < QT; ++t)
#pragma unroll
      for (int mm = 0; mm < MT; ++mm)
        acc[mm][t] = __builtin_amdgcn_wmma_f32_16x16x32_bf16(
            false, aA[t], false, bA[mm], (short)0, acc[mm][t], false, false);
    // prefetch ki+2 into A buffers (clamped to a valid addr on last iter; unused)
    int k2 = (ki + 2 < 16) ? (ki + 2) : 0;
#pragma unroll
    for (int mm = 0; mm < MT; ++mm)
      bA[mm] = *(const v16bf*)(vBase + (size_t)mm * (NH * 8192) + k2 * 512);
#pragma unroll
    for (int t = 0; t < QT; ++t)
      aA[t] = *(const v16bf*)(aBase + (size_t)t * 8192 + k2 * 512);
    // consume B buffers (ki+1)
#pragma unroll
    for (int t = 0; t < QT; ++t)
#pragma unroll
      for (int mm = 0; mm < MT; ++mm)
        acc[mm][t] = __builtin_amdgcn_wmma_f32_16x16x32_bf16(
            false, aB[t], false, bB[mm], (short)0, acc[mm][t], false, false);
  }

  {
    int colA = n * 16 + lr;                          // D layout: elem r = row r (+8 if hi)
#pragma unroll
    for (int mm = 0; mm < MT; ++mm)
#pragma unroll
      for (int t = 0; t < QT; ++t)
#pragma unroll
        for (int r = 0; r < 8; ++r)
          att[mm][t][(r + hi * 8) * 68 + colA] = f2bf(acc[mm][t][r]);
  }
  __syncthreads();

  // ---- Stage 2: out tiles = att(16x64) @ Wp col-tile j, K = 64 ----
  int j = wave;
  v16bf bw0 = *(const v16bf*)(wpP + (((j * 2 + 0) * 32 + lane) << 4));
  v16bf bw1 = *(const v16bf*)(wpP + (((j * 2 + 1) * 32 + lane) << 4));
  int col = j * 16 + lr;
  float bias = bp[col];

#pragma unroll
  for (int mm = 0; mm < MT; ++mm) {
#pragma unroll
    for (int t = 0; t < QT; ++t) {
      v8f o = {};
#pragma unroll
      for (int c = 0; c < 2; ++c) {
        union { unsigned u[8]; v16bf v; } au;
        int kb = c * 32;
#pragma unroll
        for (int i = 0; i < 8; ++i) {
          int off = (hi << 4);                       // A-layout K offsets per lane half
          au.u[i] = *(const unsigned*)&att[mm][t][lr * 68 + kb + 2 * i + off];
        }
        o = __builtin_amdgcn_wmma_f32_16x16x32_bf16(
            false, au.v, false, (c ? bw1 : bw0), (short)0, o, false, false);
      }
      size_t outBase =
          ((size_t)(b * MDIM + m0 + mm) * TDIM + ((qb * QT + t) * 16 + hi * 8)) * 64 + col;
#pragma unroll
      for (int r = 0; r < 8; ++r)
        out[outBase + (size_t)r * 64] = o[r] + bias;
    }
  }
}

// ---------------------------------------------------------------------------
extern "C" void kernel_launch(void* const* d_in, const int* in_sizes, int n_in,
                              void* d_out, int out_size, void* d_ws, size_t ws_size,
                              hipStream_t stream) {
  (void)in_sizes; (void)n_in; (void)out_size; (void)ws_size;
  const float* x  = (const float*)d_in[0];
  const float* Wq = (const float*)d_in[1];
  const float* bq = (const float*)d_in[2];
  const float* Wk = (const float*)d_in[3];
  const float* bk = (const float*)d_in[4];
  const float* Wv = (const float*)d_in[5];
  const float* bv = (const float*)d_in[6];
  const float* Wp = (const float*)d_in[7];
  const float* bp = (const float*)d_in[8];

  float* out  = (float*)d_out;                                   // [4,190,512,64]
  float* aout = out + (size_t)BS * MDIM * TDIM * CDIM;           // [4,512,512]

  char* ws = (char*)d_ws;
  float* xs = (float*)(ws + 0);                                  // 512 KB
  float* qt = (float*)(ws + 524288);                             // 512 KB
  float* kt = (float*)(ws + 1048576);                            // 512 KB
  unsigned short* wpP = (unsigned short*)(ws + 1572864);         // 8 KB
  unsigned short* wP  = (unsigned short*)(ws + 1581056);         // 8 MB
  unsigned short* vP  = (unsigned short*)(ws + 9969664);         // 49.8 MB

  hipLaunchKernelGGL(pack_wp_kernel, dim3(16), dim3(256), 0, stream, Wp, wpP);
  hipLaunchKernelGGL(reduce_x_kernel, dim3(512), dim3(256), 0, stream, x, xs);
  hipLaunchKernelGGL(qtkt_kernel, dim3(512), dim3(256), 0, stream,
                     xs, Wq, bq, Wk, bk, qt, kt);
  hipLaunchKernelGGL(vproj_pack_kernel, dim3((BS * MDIM * TDIM) / 4), dim3(256), 0, stream,
                     x, Wv, bv, vP);
  hipLaunchKernelGGL(softmax_kernel, dim3(TDIM, NH, BS), dim3(128), 0, stream,
                     qt, kt, wP, aout);
  hipLaunchKernelGGL(attn_wmma_kernel,
                     dim3(TDIM / (16 * QT), MDIM / MT, BS), dim3(128), 0, stream,
                     wP, vP, wpP, bp, out);
}
